// Memory_sup_84318797955633
// MI455X (gfx1250) — matrix operand order repaired and checked by reading, not compiled, not run on hardware
//
#include <hip/hip_runtime.h>

// Problem constants (from reference: B=16, C=256, H=W=64, M=512)
#define N_TOT   65536      // B*H*W
#define C_DIM   256
#define M_DIM   512
#define HW      4096       // H*W
#define TN      64         // rows per workgroup
#define ROWP    520        // padded LDS row stride in halves (512 + 8, keeps 16B align, breaks bank conflicts)

typedef _Float16 v8h  __attribute__((ext_vector_type(8)));
typedef _Float16 v16h __attribute__((ext_vector_type(16)));
typedef float    v8f  __attribute__((ext_vector_type(8)));

union AFrag { v16h v; v8h h[2]; };
union F8    { v8f  v; float f[8]; };

// ---------------------------------------------------------------------------
// Zero the global accumulators: colsum[512] | chan_sum[256] | chan_sumsq[256]
__global__ void k_zero_acc(float* __restrict__ acc) {
  int i = blockIdx.x * blockDim.x + threadIdx.x;
  if (i < 1024) acc[i] = 0.0f;
}

// ---------------------------------------------------------------------------
// Pack the three WMMA B-operand tables into fragment-major f16 layout.
// Fragment = 32x16 (KxN) f16 tile; per lane 16 halves (32 bytes) so the main
// kernel loads one B fragment with a single 32B global load per lane.
// B-frag lane layout (ISA 16-bit table order): lanes 0-15 hold K=0..7,16..23,
// lanes 16-31 hold K=8..15,24..31; lane n holds column n (mod 16).
//   table 0: B1 = mem^T  (K=256, N=512)  B[k][n] = mem[n][k]
//   table 1: B2 = mem    (K=512, N=256)  B[k][n] = mem[k][n]
//   table 2: B3 = conv_w^T (K=512, N=256) B[k][n] = conv_w[n][k]
__global__ void k_pack_b(const float* __restrict__ mem,
                         const float* __restrict__ conv_w,
                         _Float16* __restrict__ pb) {
  int idx = blockIdx.x * blockDim.x + threadIdx.x;
  if (idx >= 3 * 131072) return;
  int table = idx / 131072;
  int r     = idx - table * 131072;
  int f     = r >> 9;          // fragment id (512 halves each)
  int lane  = (r >> 4) & 31;
  int e     = r & 15;
  int kk    = ((lane >> 4) << 3) + (e & 7) + ((e >> 3) << 4);   // K within 32-chunk
  int ncol  = lane & 15;
  float val;
  if (table == 0) {
    int kc = f >> 5, nc = f & 31;
    val = mem[(nc * 16 + ncol) * C_DIM + (kc * 32 + kk)];
  } else if (table == 1) {
    int kc = f >> 4, nc = f & 15;
    val = mem[(kc * 32 + kk) * C_DIM + (nc * 16 + ncol)];
  } else {
    int kc = f >> 4, nc = f & 15;
    val = conv_w[(nc * 16 + ncol) * (2 * C_DIM) + (kc * 32 + kk)];
  }
  pb[idx] = (_Float16)val;
}

// ---------------------------------------------------------------------------
// Fused main kernel: 128 threads = 4 wave32, 64 query rows per block.
// Each wave owns a 16-row strip and does all three WMMA GEMMs for it.
__global__ __launch_bounds__(128)
void k_fused(const float* __restrict__ query,
             const _Float16* __restrict__ pb,
             float* __restrict__ out_y,      // y_pre staged in final [B,O,H,W] layout
             float* __restrict__ out_ssq,    // raw exp(score); rescaled by k_scale_cols
             float* __restrict__ out_ssm,    // exp(score)/rowsum
             float* __restrict__ g_colsum,
             float* __restrict__ g_csum,
             float* __restrict__ g_cssq) {
  extern __shared__ char smem[];
  _Float16* upd  = (_Float16*)smem;          // [TN][ROWP] f16: cols 0..255 = q, 256..511 = concat_memory
  _Float16* Eb   = upd + TN * ROWP;          // [TN][ROWP] f16: exp(score)
  float* rowsum  = (float*)(Eb + TN * ROWP); // [TN]
  float* colpart = rowsum + TN;              // [512]
  float* csumL   = colpart + M_DIM;          // [256]
  float* cssqL   = csumL + C_DIM;            // [256]
  float* normtmp = cssqL + C_DIM;            // [128]
  float* trbase  = normtmp + 128;            // 4 waves * [16][17]

  const int tid  = threadIdx.x;
  const int lane = tid & 31;
  const int wid  = tid >> 5;
  const int n0   = blockIdx.x * TN;
  const int b    = n0 >> 12;                 // HW = 4096 = 2^12
  const int hw0  = n0 & (HW - 1);

  const int rbase  = wid * 16;
  const int rowsel = (lane & 16) ? 8 : 0;    // C/D row offset for this half-wave
  const int mcol   = lane & 15;
  const int koffA  = (lane & 16) ? 8 : 0;    // A-frag K offset (ISA 16-bit A layout)

  const v16h* pb1 = (const v16h*)pb;         // 256 frags * 32 lanes
  const v16h* pb2 = pb1 + 256 * 32;
  const v16h* pb3 = pb2 + 256 * 32;

  // zero LDS accumulators
  if (tid < TN) rowsum[tid] = 0.0f;
  for (int i = tid; i < M_DIM; i += 128) colpart[i] = 0.0f;
  for (int i = tid; i < C_DIM; i += 128) { csumL[i] = 0.0f; cssqL[i] = 0.0f; }

  // ---- Phase 1: channel-normalize query -> upd[:, 0:256] (f16) --------------
  {
    int r = tid >> 1, half = tid & 1;
    int n = n0 + r;
    const float* qp = query + (size_t)b * C_DIM * HW + (n & (HW - 1));
    float ss = 0.0f;
    int c0 = half * 128;
    for (int c = c0; c < c0 + 128; ++c) {
      float v = qp[(size_t)c * HW];
      ss += v * v;
      upd[r * ROWP + c] = (_Float16)v;
    }
    normtmp[tid] = ss;
    __syncthreads();
    float tot = normtmp[tid & ~1] + normtmp[tid | 1];
    float inv = 1.0f / fmaxf(sqrtf(tot), 1e-12f);
    for (int c = c0; c < c0 + 128; ++c)
      upd[r * ROWP + c] = (_Float16)((float)upd[r * ROWP + c] * inv);
  }
  __syncthreads();

  // ---- Phase 2: score = q @ mem^T via WMMA, exp, row/col sums ---------------
  float rowacc[8];
#pragma unroll
  for (int v = 0; v < 8; ++v) rowacc[v] = 0.0f;

  for (int nc = 0; nc < 32; ++nc) {          // 512 memory slots / 16
    F8 d;
#pragma unroll
    for (int v = 0; v < 8; ++v) d.f[v] = 0.0f;
    for (int kc = 0; kc < 8; ++kc) {         // K = 256 / 32
      const _Float16* arow = upd + (rbase + mcol) * ROWP + kc * 32 + koffA;
      AFrag a;
      a.h[0] = *(const v8h*)(arow);
      a.h[1] = *(const v8h*)(arow + 16);
      v16h bf = pb1[(kc * 32 + nc) * 32 + lane];
      d.v = __builtin_amdgcn_wmma_f32_16x16x32_f16(false, a.v, false, bf,
                                                   (short)0, d.v, false, false);
    }
    // |score| <= 1 (unit vectors) -> exp without max subtraction is safe
    float cacc = 0.0f;
    int mg = nc * 16 + mcol;
#pragma unroll
    for (int v = 0; v < 8; ++v) {
      float e = __expf(d.f[v]);
      rowacc[v] += e;
      cacc += e;
      int rl = rbase + rowsel + v;
      out_ssq[(size_t)(n0 + rl) * M_DIM + mg] = e;     // raw exp, scaled later
      Eb[rl * ROWP + mg] = (_Float16)e;
    }
    atomicAdd(colpart + mg, cacc);
  }
#pragma unroll
  for (int v = 0; v < 8; ++v)
    atomicAdd(rowsum + rbase + rowsel + v, rowacc[v]);
  __syncthreads();

  for (int i = tid; i < M_DIM; i += 128)
    atomicAdd(g_colsum + i, colpart[i]);

  // ---- Phase 3: concat_memory = (E/rowsum) @ mem via WMMA -------------------
  float invr[8];
#pragma unroll
  for (int v = 0; v < 8; ++v) invr[v] = 1.0f / rowsum[rbase + rowsel + v];

  for (int nc = 0; nc < 16; ++nc) {          // C = 256 / 16
    F8 d;
#pragma unroll
    for (int v = 0; v < 8; ++v) d.f[v] = 0.0f;
    for (int kc = 0; kc < 16; ++kc) {        // K = 512 / 32
      const _Float16* arow = Eb + (rbase + mcol) * ROWP + kc * 32 + koffA;
      AFrag a;
      a.h[0] = *(const v8h*)(arow);
      a.h[1] = *(const v8h*)(arow + 16);
      v16h bf = pb2[(kc * 16 + nc) * 32 + lane];
      d.v = __builtin_amdgcn_wmma_f32_16x16x32_f16(false, a.v, false, bf,
                                                   (short)0, d.v, false, false);
    }
#pragma unroll
    for (int v = 0; v < 8; ++v) {
      int rl = rbase + rowsel + v;
      upd[rl * ROWP + 256 + nc * 16 + mcol] = (_Float16)(d.f[v] * invr[v]);
    }
  }

  // row-softmax output (coalesced)
  for (int idx = tid; idx < TN * M_DIM; idx += 128) {
    int r = idx >> 9, m = idx & (M_DIM - 1);
    out_ssm[(size_t)(n0 + r) * M_DIM + m] = (float)Eb[r * ROWP + m] / rowsum[r];
  }
  __syncthreads();

  // ---- Phase 4: y_pre = updated @ conv_w^T via WMMA; BN stats; store --------
  float* trb = trbase + wid * (16 * 17);
  for (int nc = 0; nc < 16; ++nc) {          // O = 256 / 16
    F8 d;
#pragma unroll
    for (int v = 0; v < 8; ++v) d.f[v] = 0.0f;
    for (int kc = 0; kc < 16; ++kc) {        // K = 512 / 32
      const _Float16* arow = upd + (rbase + mcol) * ROWP + kc * 32 + koffA;
      AFrag a;
      a.h[0] = *(const v8h*)(arow);
      a.h[1] = *(const v8h*)(arow + 16);
      v16h bf = pb3[(kc * 16 + nc) * 32 + lane];
      d.v = __builtin_amdgcn_wmma_f32_16x16x32_f16(false, a.v, false, bf,
                                                   (short)0, d.v, false, false);
    }
    float s = 0.0f, s2 = 0.0f;
#pragma unroll
    for (int v = 0; v < 8; ++v) {
      float y = d.f[v];
      s += y; s2 += y * y;
      trb[(rowsel + v) * 17 + mcol] = y;     // transpose staging, padded stride
    }
    int o = nc * 16 + mcol;
    atomicAdd(csumL + o, s);
    atomicAdd(cssqL + o, s2);
    __syncthreads();                         // uniform across all 4 waves
    int nloc = lane & 15;
    int og   = lane >> 4;
#pragma unroll
    for (int i = 0; i < 8; ++i) {
      int ol = og + i * 2;
      float val = trb[nloc * 17 + ol];
      int oglob = nc * 16 + ol;
      // coalesced along hw in final [B, O, H, W] layout
      out_y[(size_t)b * (C_DIM * HW) + (size_t)oglob * HW + hw0 + rbase + nloc] = val;
    }
    __syncthreads();
  }

  for (int i = tid; i < C_DIM; i += 128) {
    atomicAdd(g_csum + i, csumL[i]);
    atomicAdd(g_cssq + i, cssqL[i]);
  }
}

// ---------------------------------------------------------------------------
// In-place: softmax over queries (axis 0): E[n][m] / colsum[m]
__global__ void k_scale_cols(float* __restrict__ ssq,
                             const float* __restrict__ colsum) {
  size_t i = (size_t)blockIdx.x * blockDim.x + threadIdx.x;
  int m = (int)(i & (M_DIM - 1));
  ssq[i] = ssq[i] / colsum[m];
}

// ---------------------------------------------------------------------------
// In-place BN (training-mode batch stats) + ReLU on y
__global__ void k_bn_relu(float* __restrict__ y,
                          const float* __restrict__ csum,
                          const float* __restrict__ cssq,
                          const float* __restrict__ gamma,
                          const float* __restrict__ beta) {
  size_t i = (size_t)blockIdx.x * blockDim.x + threadIdx.x;
  int o = (int)((i >> 12) & (C_DIM - 1));
  const float invN = 1.0f / (float)N_TOT;
  float mean = csum[o] * invN;
  float var  = cssq[o] * invN - mean * mean;
  float sc   = gamma[o] * rsqrtf(var + 1e-5f);
  float v = (y[i] - mean) * sc + beta[o];
  y[i] = fmaxf(v, 0.0f);
}

// ---------------------------------------------------------------------------
extern "C" void kernel_launch(void* const* d_in, const int* in_sizes, int n_in,
                              void* d_out, int out_size, void* d_ws, size_t ws_size,
                              hipStream_t stream) {
  const float* query  = (const float*)d_in[0];
  const float* mem    = (const float*)d_in[1];
  const float* conv_w = (const float*)d_in[2];
  const float* gamma  = (const float*)d_in[3];
  const float* beta   = (const float*)d_in[4];

  float* out      = (float*)d_out;
  float* out_y    = out;                                   // 16*256*64*64
  float* out_ssq  = out + (size_t)16777216;                // [N, M]
  float* out_ssm  = out + (size_t)16777216 + 33554432;     // [N, M]

  char* ws = (char*)d_ws;
  _Float16* pb = (_Float16*)ws;                 // 3 * 131072 halves = 768 KB
  float* acc   = (float*)(ws + 786432);         // colsum[512] | csum[256] | cssq[256]
  float* colsum = acc;
  float* csum   = acc + 512;
  float* cssq   = acc + 768;

  k_zero_acc<<<1, 1024, 0, stream>>>(acc);
  k_pack_b<<<(3 * 131072 + 255) / 256, 256, 0, stream>>>(mem, conv_w, pb);

  size_t smem = (size_t)(2 * TN * ROWP * sizeof(_Float16))
              + sizeof(float) * (TN + M_DIM + C_DIM + C_DIM + 128 + 4 * 16 * 17);
  k_fused<<<N_TOT / TN, 128, smem, stream>>>(query, pb, out_y, out_ssq, out_ssm,
                                             colsum, csum, cssq);

  k_scale_cols<<<(N_TOT * (size_t)M_DIM) / 256, 256, 0, stream>>>(out_ssq, colsum);
  k_bn_relu<<<16777216 / 256, 256, 0, stream>>>(out_y, csum, cssq, gamma, beta);
}
